// LSTM_3298534883859
// MI455X (gfx1250) — matrix operand order repaired and checked by reading, not compiled, not run on hardware
//
#include <hip/hip_runtime.h>
#include <hip/hip_bf16.h>

// ---------------------------------------------------------------------------
// Problem constants (from reference)
// ---------------------------------------------------------------------------
#define B_   256
#define T_   128
#define H_   1024
#define Z_   1026   // STATE_DIM + HIDDEN
#define INCELL_ 1027
#define NG_  4096   // 4*HIDDEN

// d_out section offsets (flat floats, return order: outputs, z, c_z, coeffs, mats)
#define OUT_OUTPUTS 0
#define OUT_Z       ((size_t)B_ * T_ * Z_)                 // 33,619,968
#define OUT_CZ      (OUT_Z + (size_t)B_ * Z_)              // 33,882,624
#define OUT_COEFF   (OUT_CZ + (size_t)B_ * Z_)             // 34,145,280
#define OUT_MATS    (OUT_COEFF + (size_t)B_ * T_ * 2)      // 34,210,816

// workspace byte offsets
#define WS_BW   0                         // packed weights bf16: 4096*1024*2 = 8 MiB
#define WS_A0   (8u*1024u*1024u)          // A-frag ping: 256*1024*2 = 512 KiB
#define WS_A1   (WS_A0 + 512u*1024u)      // A-frag pong
#define WS_C    (WS_A1 + 512u*1024u)      // c state f32: 256*1024*4 = 1 MiB
#define WS_X    (WS_C + 1024u*1024u)      // x state f32: 256*2*4 = 2 KiB

typedef __attribute__((ext_vector_type(16))) __bf16 v16bf;
typedef __attribute__((ext_vector_type(8)))  float  v8f;

__device__ __forceinline__ __bf16 f2bf(float f) {
    union { float f; unsigned u; } in; in.f = f;
    unsigned u = in.u;
    unsigned r = u + 0x7FFFu + ((u >> 16) & 1u);   // round-to-nearest-even
    union { unsigned short s; __bf16 b; } out;
    out.s = (unsigned short)(r >> 16);
    return out.b;
}

__device__ __forceinline__ float sigm(float x) {
    return 1.0f / (1.0f + __expf(-x));
}

// ---------------------------------------------------------------------------
// Pack WU_w hidden columns (cols 3..1026) into WMMA B-fragment order (bf16).
// B tile (gate g, col-tile ct, k-chunk kc): lane holds column N = ct*16 + lane%16,
// K = kc*32 + (j/8)*16 + 8*(lane/16) + (j%8)  (j = 0..15, matches A-fragment order).
// flat packed index = (((g*64+ct)*32+kc)*32+lane)*16 + j
// ---------------------------------------------------------------------------
__global__ void __launch_bounds__(256) pack_weights_kernel(
    const float* __restrict__ WU_w, __bf16* __restrict__ Bw)
{
    unsigned idx = blockIdx.x * 256u + threadIdx.x;   // 0 .. 4096*1024-1
    int j    =  idx        & 15;
    int lane = (idx >> 4)  & 31;
    int kc   = (idx >> 9)  & 31;
    int ct   = (idx >> 14) & 63;
    int g    =  idx >> 20;
    int N   = ct * 16 + (lane & 15);
    int K   = kc * 32 + (j >> 3) * 16 + (lane >> 4) * 8 + (j & 7);
    int row = g * H_ + N;
    Bw[idx] = f2bf(WU_w[(size_t)row * INCELL_ + 3 + K]);
}

// ---------------------------------------------------------------------------
// Initial state: pack h0 (from z0) into A-fragment layout, copy c0 and x0.
// A tile (row-tile rt, k-chunk kc): lane = (m=lane%16, half=lane/16),
// element j holds K = kc*32 + (j/8)*16 + half*8 + (j%8).
// ---------------------------------------------------------------------------
__global__ void __launch_bounds__(256) init_state_kernel(
    const float* __restrict__ z0, const float* __restrict__ c_z0,
    __bf16* __restrict__ A0, float* __restrict__ c_ws, float* __restrict__ x_ws)
{
    unsigned idx = blockIdx.x * 256u + threadIdx.x;   // 0 .. 256*1024-1
    // --- h0 pack ---
    {
        int j    =  idx        & 15;
        int lane = (idx >> 4)  & 31;
        int kc   = (idx >> 9)  & 31;
        int rt   =  idx >> 14;
        int m    = lane & 15;
        int half = lane >> 4;
        int K    = kc * 32 + (j >> 3) * 16 + half * 8 + (j & 7);
        int b    = rt * 16 + m;
        A0[idx]  = f2bf(z0[(size_t)b * Z_ + 2 + K]);
    }
    // --- c0 copy ---
    {
        int b = idx >> 10;
        int k = idx & 1023;
        c_ws[idx] = c_z0[(size_t)b * Z_ + 2 + k];
    }
    // --- x0 copy ---
    if (idx < 2u * B_) {
        x_ws[idx] = z0[(size_t)(idx >> 1) * Z_ + (idx & 1)];
    }
}

// ---------------------------------------------------------------------------
// Per-step fused GEMM + LSTM cell, split-K for short critical path.
//
// Block = 256 threads = 8 waves = 2 batch-tiles x 4 K-slices (K=256 each).
// All 8 waves share the same hidden-col tile `ct` -> waves (w, w+4) issue
// identical B-fragment addresses (WGP$ reuse). Each wave does 8 k-chunks x
// 4 gates of v_wmma_f32_16x16x32_bf16, partials are reduced through LDS,
// then the s==0 wave of each tile applies the LSTM nonlinearity and writes
// h_new (f32 -> outputs, bf16 -> next-step A fragments) and c_new in place.
// ---------------------------------------------------------------------------
__global__ void __launch_bounds__(256) gemm_lstm_kernel(
    const __bf16* __restrict__ Aread, __bf16* __restrict__ Awrite,
    const __bf16* __restrict__ Bw,
    float* __restrict__ c_ws, const float* __restrict__ x_ws,
    const float* __restrict__ WU_w, const float* __restrict__ WU_b,
    const float* __restrict__ rnn_input,
    float* __restrict__ out, int t)
{
    const int lane = threadIdx.x & 31;
    const int w    = threadIdx.x >> 5;       // 0..7
    const int tile = w >> 2;                 // 0..1  (batch tile within block)
    const int s    = w & 3;                  // K-slice 0..3
    const int ct   = blockIdx.x & 63;        // hidden-col tile
    const int rt   = (blockIdx.x >> 6) * 2 + tile;   // batch tile 0..15

    const v16bf* __restrict__ Av = reinterpret_cast<const v16bf*>(Aread);
    const v16bf* __restrict__ Bv = reinterpret_cast<const v16bf*>(Bw);

    v8f acc[4] = {{0.f}, {0.f}, {0.f}, {0.f}};
    const int kc0   = s * 8;
    const int abase = rt * 1024 + lane;      // (rt*32+kc)*32+lane
    const int bbase = ct * 1024 + lane;      // ((g*64+ct)*32+kc)*32+lane
    const int gstr  = 64 * 32 * 32;          // per-gate stride (v16bf units)

#pragma unroll
    for (int i = 0; i < 8; ++i) {
        const int kc = kc0 + i;
        v16bf a = Av[abase + kc * 32];
        v16bf bfrag[4];
#pragma unroll
        for (int g = 0; g < 4; ++g)
            bfrag[g] = Bv[bbase + kc * 32 + g * gstr];
#pragma unroll
        for (int g = 0; g < 4; ++g)
            acc[g] = __builtin_amdgcn_wmma_f32_16x16x32_bf16(
                false, a, false, bfrag[g], (short)0, acc[g], false, false);
    }

    // ---- split-K reduction through LDS (conflict-free layout) ----
    // red[tile][(g*8+r)*4 + s][lane]
    __shared__ float red[2 * 4096];
    float* rr = red + tile * 4096;
#pragma unroll
    for (int g = 0; g < 4; ++g)
#pragma unroll
        for (int r = 0; r < 8; ++r)
            rr[((g * 8 + r) * 4 + s) * 32 + lane] = acc[g][r];
    __syncthreads();

    if (s == 0) {
        float gsum[4][8];
#pragma unroll
        for (int g = 0; g < 4; ++g)
#pragma unroll
            for (int r = 0; r < 8; ++r) {
                const float* p = rr + (g * 8 + r) * 128 + lane;
                gsum[g][r] = p[0] + p[32] + p[64] + p[96];
            }

        // ---- rank-1 (x1,x2,u) + bias terms, then LSTM cell ----
        const int n     = lane & 15;
        const int halfl = lane >> 4;
        const int col   = ct * 16 + n;

        float w0[4], w1[4], w2[4], bb[4];
#pragma unroll
        for (int g = 0; g < 4; ++g) {
            int row = g * H_ + col;
            const float* wr = WU_w + (size_t)row * INCELL_;
            w0[g] = wr[0]; w1[g] = wr[1]; w2[g] = wr[2];
            bb[g] = WU_b[row];
        }

        const int j2_base = (ct & 1) * 8 + (n & 7);  // target elem in next A frag
        const int kc2     = ct >> 1;
        const int lane2hi = (n >> 3) * 16;

#pragma unroll
        for (int r = 0; r < 8; ++r) {
            int m = r + 8 * halfl;
            int b = rt * 16 + m;
            float x1 = x_ws[b * 2 + 0];
            float x2 = x_ws[b * 2 + 1];
            float u  = rnn_input[(size_t)b * T_ + t];

            float gi = gsum[0][r] + x1 * w0[0] + x2 * w1[0] + u * w2[0] + bb[0];
            float gf = gsum[1][r] + x1 * w0[1] + x2 * w1[1] + u * w2[1] + bb[1];
            float gg = gsum[2][r] + x1 * w0[2] + x2 * w1[2] + u * w2[2] + bb[2];
            float go = gsum[3][r] + x1 * w0[3] + x2 * w1[3] + u * w2[3] + bb[3];

            size_t cidx = (size_t)b * H_ + col;
            float c0v = c_ws[cidx];
            float cn  = sigm(gf) * c0v + sigm(gi) * tanhf(gg);
            float hn  = sigm(go) * tanhf(cn);
            c_ws[cidx] = cn;

            out[OUT_OUTPUTS + ((size_t)b * T_ + t) * Z_ + 2 + col] = hn;

            // scatter h_new (bf16) into next-step A-fragment layout
            int lane2 = lane2hi + m;
            Awrite[(((size_t)(rt * 32 + kc2) * 32 + lane2) << 4) + j2_base] = f2bf(hn);
        }
    }
}

// ---------------------------------------------------------------------------
// Per-step state update: alpha/cand projections (dot over H), VdP dynamics,
// x_next, coefficients, mats. One block per batch element.
// ---------------------------------------------------------------------------
__global__ void __launch_bounds__(128) state_update_kernel(
    float* __restrict__ out,
    const float* __restrict__ alpha_w, const float* __restrict__ alpha_b,
    const float* __restrict__ Wx_w,    const float* __restrict__ Wx_b,
    const float* __restrict__ rnn_input, const float* __restrict__ tau,
    float* __restrict__ x_ws, int t)
{
    const int b   = blockIdx.x;
    const int tid = threadIdx.x;
    const float* h = out + OUT_OUTPUTS + ((size_t)b * T_ + t) * Z_ + 2;

    float s0 = 0.f, s1 = 0.f, s2 = 0.f, s3 = 0.f;
    for (int k = tid; k < H_; k += 128) {
        float hv = h[k];
        s0 += hv * alpha_w[k];
        s1 += hv * alpha_w[H_ + k];
        s2 += hv * Wx_w[k];
        s3 += hv * Wx_w[H_ + k];
    }
    __shared__ float red[512];
    red[tid]       = s0;
    red[128 + tid] = s1;
    red[256 + tid] = s2;
    red[384 + tid] = s3;
    __syncthreads();
    for (int off = 64; off > 0; off >>= 1) {
        if (tid < off) {
            red[tid]       += red[tid + off];
            red[128 + tid] += red[128 + tid + off];
            red[256 + tid] += red[256 + tid + off];
            red[384 + tid] += red[384 + tid + off];
        }
        __syncthreads();
    }
    if (tid == 0) {
        float a0 = sigm(red[0]   + alpha_b[0]);
        float a1 = sigm(red[128] + alpha_b[1]);
        float cand0 = red[256] + Wx_b[0];
        float cand1 = red[384] + Wx_b[1];

        float x1 = x_ws[b * 2 + 0];
        float x2 = x_ws[b * 2 + 1];
        float u  = rnn_input[(size_t)b * T_ + t];
        float tv = tau[(size_t)b * T_ + t];

        // MU = 1, DYN_FACTOR = 1
        float a21 = -1.0f - 2.0f * x1 * x2;
        float a22 = 1.0f - x1 * x1;
        float xdot0 = x2;                                     // [0 1]·x + 0*u + 0
        float xdot1 = a21 * x1 + a22 * x2 + u + 2.0f * x1 * x1 * x2;
        float xm0 = x1 + tv * xdot0;
        float xm1 = x2 + tv * xdot1;
        float xn0 = a0 * xm0 + (1.0f - a0) * cand0;
        float xn1 = a1 * xm1 + (1.0f - a1) * cand1;

        size_t obase = ((size_t)b * T_ + t) * Z_;
        out[OUT_OUTPUTS + obase + 0] = xn0;
        out[OUT_OUTPUTS + obase + 1] = xn1;
        out[OUT_COEFF + ((size_t)b * T_ + t) * 2 + 0] = a0;
        out[OUT_COEFF + ((size_t)b * T_ + t) * 2 + 1] = a1;
        x_ws[b * 2 + 0] = xn0;
        x_ws[b * 2 + 1] = xn1;
        if (b == 0) {
            out[OUT_MATS + (size_t)t * 4 + 0] = 0.0f;
            out[OUT_MATS + (size_t)t * 4 + 1] = 1.0f;
            out[OUT_MATS + (size_t)t * 4 + 2] = a21;
            out[OUT_MATS + (size_t)t * 4 + 3] = a22;
        }
    }
}

// ---------------------------------------------------------------------------
// Finalize: z = last-step outputs; c_z = [c_x from c_z0, final c_new].
// ---------------------------------------------------------------------------
__global__ void __launch_bounds__(256) finalize_kernel(
    float* __restrict__ out, const float* __restrict__ c_z0,
    const float* __restrict__ c_ws)
{
    unsigned idx = blockIdx.x * 256u + threadIdx.x;   // 0 .. 256*1026-1
    if (idx >= (unsigned)(B_ * Z_)) return;
    int b = idx / Z_;
    int k = idx % Z_;
    out[OUT_Z + idx] = out[OUT_OUTPUTS + ((size_t)b * T_ + (T_ - 1)) * Z_ + k];
    out[OUT_CZ + idx] = (k < 2) ? c_z0[idx] : c_ws[(size_t)b * H_ + (k - 2)];
}

// ---------------------------------------------------------------------------
extern "C" void kernel_launch(void* const* d_in, const int* in_sizes, int n_in,
                              void* d_out, int out_size, void* d_ws, size_t ws_size,
                              hipStream_t stream)
{
    const float* rnn_input = (const float*)d_in[0];
    const float* z0        = (const float*)d_in[1];
    const float* c_z0      = (const float*)d_in[2];
    const float* tau       = (const float*)d_in[3];
    const float* WU_w      = (const float*)d_in[4];
    const float* WU_b      = (const float*)d_in[5];
    const float* alpha_w   = (const float*)d_in[6];
    const float* alpha_b   = (const float*)d_in[7];
    const float* Wx_w      = (const float*)d_in[8];
    const float* Wx_b      = (const float*)d_in[9];

    char* ws = (char*)d_ws;
    __bf16* Bw   = (__bf16*)(ws + WS_BW);
    __bf16* A0   = (__bf16*)(ws + WS_A0);
    __bf16* A1   = (__bf16*)(ws + WS_A1);
    float*  c_ws = (float*)(ws + WS_C);
    float*  x_ws = (float*)(ws + WS_X);
    float*  out  = (float*)d_out;

    // one-time per launch (deterministic): pack weights + initial state
    pack_weights_kernel<<<(NG_ * H_) / 256, 256, 0, stream>>>(WU_w, Bw);
    init_state_kernel<<<(B_ * H_) / 256, 256, 0, stream>>>(z0, c_z0, A0, c_ws, x_ws);

    for (int t = 0; t < T_; ++t) {
        const __bf16* Ar = (t & 1) ? A1 : A0;
        __bf16*       Aw = (t & 1) ? A0 : A1;
        // 512 blocks x 8 waves = 4096 waves (16 rt x 64 ct x 4 K-slices)
        gemm_lstm_kernel<<<512, 256, 0, stream>>>(Ar, Aw, Bw, c_ws, x_ws,
                                                  WU_w, WU_b, rnn_input, out, t);
        state_update_kernel<<<256, 128, 0, stream>>>(out, alpha_w, alpha_b,
                                                     Wx_w, Wx_b, rnn_input, tau,
                                                     x_ws, t);
    }

    finalize_kernel<<<(B_ * Z_ + 255) / 256, 256, 0, stream>>>(out, c_z0, c_ws);
}
